// Attention_11733850652898
// MI455X (gfx1250) — matrix operand order repaired and checked by reading, not compiled, not run on hardware
//
#include <hip/hip_runtime.h>
#include <hip/hip_bf16.h>
#include <math.h>

// ---------------------------------------------------------------------------
// Complex attention (B=4, N=1024, D=1024, H=16, DH=64) for gfx1250 (MI455X).
// All four GEMM stages run on v_wmma_f32_16x16x32_bf16 (f32 accumulate).
// Attention is flash-style (no [B,H,N,N] materialization); row softmax stats
// are reduced with ds_swizzle_b32 butterflies inside each 16-lane group.
// The QKV GEMM emits both row-major w (for Q/K fragments) and transposed wT
// (for contiguous V B-fragments in the PV product).
// ---------------------------------------------------------------------------

typedef __attribute__((ext_vector_type(16))) __bf16        v16bf;
typedef __attribute__((ext_vector_type(8)))  __bf16        v8bf;
typedef __attribute__((ext_vector_type(8)))  float         v8f;
typedef __attribute__((ext_vector_type(8)))  unsigned int  v8u;

static constexpr int   Bz = 4, Nz = 1024, Dz = 1024, Hz = 16, DHz = 64;
static constexpr int   INNERz = Hz * DHz;          // 1024
static constexpr int   Mrows  = Bz * Nz;           // 4096
static constexpr float SCALEz = 0.125f;            // DH^-0.5

// ---- WMMA helper: D = A(16x32 bf16) * B(32x16 bf16) + C(f32) --------------
static __device__ __forceinline__ v8f wmma_bf(v16bf a, v16bf b, v8f c) {
  return __builtin_amdgcn_wmma_f32_16x16x32_bf16(
      /*neg_a=*/false, a, /*neg_b=*/false, b,
      /*c_mod=*/(short)0, c, /*reuse_a=*/false, /*reuse_b=*/false);
}

// A-fragment (16x32, row-major in memory). Lane holds row lane%16.
// K(e,g): e<8 -> g*8+e ; e>=8 -> 16+g*8+(e-8)  => two contiguous 16B loads.
static __device__ __forceinline__ v16bf ldfragA(const __bf16* p, int g) {
  const v8bf lo = *(const v8bf*)(p + 8 * g);
  const v8bf hi = *(const v8bf*)(p + 16 + 8 * g);
  return __builtin_shufflevector(lo, hi, 0, 1, 2, 3, 4, 5, 6, 7,
                                         8, 9, 10, 11, 12, 13, 14, 15);
}

// B-fragment (32x16). Lane holds column lane%16, K(e,g) same as A fragment.
static __device__ __forceinline__ v16bf ldfragB(const __bf16* p, int g) {
  const v8bf lo = *(const v8bf*)(p + 8 * g);
  const v8bf hi = *(const v8bf*)(p + 16 + 8 * g);
  return __builtin_shufflevector(lo, hi, 0, 1, 2, 3, 4, 5, 6, 7,
                                         8, 9, 10, 11, 12, 13, 14, 15);
}

// Negate a packed bf16 fragment (flip sign bits).
static __device__ __forceinline__ v16bf negbf(v16bf x) {
  v8u u = __builtin_bit_cast(v8u, x);
  const v8u m = {0x80008000u, 0x80008000u, 0x80008000u, 0x80008000u,
                 0x80008000u, 0x80008000u, 0x80008000u, 0x80008000u};
  u = u ^ m;
  return __builtin_bit_cast(v16bf, u);
}

// ---- 16-lane butterfly reductions via ds_swizzle (group-of-32 mode) -------
#define SWZ(x, imm)                                                        \
  __builtin_bit_cast(float, __builtin_amdgcn_ds_swizzle(                   \
                                __builtin_bit_cast(int, (x)), (imm)))

static __device__ __forceinline__ float red_max16(float x) {
  x = fmaxf(x, SWZ(x, 0x041F));   // xor 1
  x = fmaxf(x, SWZ(x, 0x081F));   // xor 2
  x = fmaxf(x, SWZ(x, 0x101F));   // xor 4
  x = fmaxf(x, SWZ(x, 0x201F));   // xor 8
  return x;
}
static __device__ __forceinline__ float red_sum16(float x) {
  x += SWZ(x, 0x041F);
  x += SWZ(x, 0x081F);
  x += SWZ(x, 0x101F);
  x += SWZ(x, 0x201F);
  return x;
}

// ---------------------------------------------------------------------------
// f32 -> bf16 convert (grid-stride)
// ---------------------------------------------------------------------------
__global__ void cvt_f32_bf16_kernel(const float* __restrict__ in,
                                    __bf16* __restrict__ out, int n) {
  int i = blockIdx.x * blockDim.x + threadIdx.x;
  const int stride = gridDim.x * blockDim.x;
  for (; i < n; i += stride) out[i] = (__bf16)in[i];
}

// ---------------------------------------------------------------------------
// Complex GEMM: C = A * B^T, A [M,K], B [N,K], all planar re/im bf16.
// One 16x16 C tile per wave, 8 waves/block. Writes row-major C and a
// transposed copy CT [Nn, M] (consumed as contiguous V fragments later).
// ---------------------------------------------------------------------------
__global__ __launch_bounds__(256) void gemm_cplx_bf16_kernel(
    const __bf16* __restrict__ Ar, const __bf16* __restrict__ Ai,
    const __bf16* __restrict__ Br, const __bf16* __restrict__ Bi,
    __bf16* __restrict__ Cr, __bf16* __restrict__ Ci,
    __bf16* __restrict__ CrT, __bf16* __restrict__ CiT,
    int M, int Nn, int K) {
  const int lane = threadIdx.x & 31, wave = threadIdx.x >> 5;
  const int ml = lane & 15, g = lane >> 4;
  const int tiles_n = Nn >> 4;
  const int tile = blockIdx.x * 8 + wave;
  const int tm = tile / tiles_n, tn = tile % tiles_n;
  if (tm * 16 >= M) return;  // wave-uniform: EXEC stays full for WMMA

  const __bf16* ar = Ar + (size_t)(tm * 16 + ml) * K;
  const __bf16* ai = Ai + (size_t)(tm * 16 + ml) * K;
  const __bf16* br = Br + (size_t)(tn * 16 + ml) * K;
  const __bf16* bi = Bi + (size_t)(tn * 16 + ml) * K;

  v8f c1 = {}, c2 = {}, c3 = {}, c4 = {};
  for (int k0 = 0; k0 < K; k0 += 32) {
    const v16bf a_r = ldfragA(ar + k0, g);
    const v16bf a_i = ldfragA(ai + k0, g);
    const v16bf b_r = ldfragB(br + k0, g);
    const v16bf b_i = ldfragB(bi + k0, g);
    c1 = wmma_bf(a_r, b_r, c1);   // re+
    c2 = wmma_bf(a_i, b_i, c2);   // re-
    c3 = wmma_bf(a_r, b_i, c3);   // im+
    c4 = wmma_bf(a_i, b_r, c4);   // im+
  }
#pragma unroll
  for (int r = 0; r < 8; ++r) {
    const int row = tm * 16 + r + 8 * g;   // C layout: M = r + 8*(lane/16)
    const int col = tn * 16 + ml;          // N = lane%16
    const __bf16 re = (__bf16)(c1[r] - c2[r]);
    const __bf16 im = (__bf16)(c3[r] + c4[r]);
    const size_t idx = (size_t)row * Nn + col;
    Cr[idx] = re;
    Ci[idx] = im;
    const size_t idxT = (size_t)col * M + row;
    CrT[idxT] = re;
    CiT[idxT] = im;
  }
}

// ---------------------------------------------------------------------------
// Complex GEMM writing interleaved complex64 output + bias (final projection)
// ---------------------------------------------------------------------------
__global__ __launch_bounds__(256) void gemm_cplx_out_kernel(
    const __bf16* __restrict__ Ar, const __bf16* __restrict__ Ai,
    const __bf16* __restrict__ Br, const __bf16* __restrict__ Bi,
    const float* __restrict__ bre, const float* __restrict__ bim,
    float* __restrict__ out, int M, int Nn, int K) {
  const int lane = threadIdx.x & 31, wave = threadIdx.x >> 5;
  const int ml = lane & 15, g = lane >> 4;
  const int tiles_n = Nn >> 4;
  const int tile = blockIdx.x * 8 + wave;
  const int tm = tile / tiles_n, tn = tile % tiles_n;
  if (tm * 16 >= M) return;

  const __bf16* ar = Ar + (size_t)(tm * 16 + ml) * K;
  const __bf16* ai = Ai + (size_t)(tm * 16 + ml) * K;
  const __bf16* br = Br + (size_t)(tn * 16 + ml) * K;
  const __bf16* bi = Bi + (size_t)(tn * 16 + ml) * K;

  v8f c1 = {}, c2 = {}, c3 = {}, c4 = {};
  for (int k0 = 0; k0 < K; k0 += 32) {
    const v16bf a_r = ldfragA(ar + k0, g);
    const v16bf a_i = ldfragA(ai + k0, g);
    const v16bf b_r = ldfragB(br + k0, g);
    const v16bf b_i = ldfragB(bi + k0, g);
    c1 = wmma_bf(a_r, b_r, c1);
    c2 = wmma_bf(a_i, b_i, c2);
    c3 = wmma_bf(a_r, b_i, c3);
    c4 = wmma_bf(a_i, b_r, c4);
  }
#pragma unroll
  for (int r = 0; r < 8; ++r) {
    const int row = tm * 16 + r + 8 * g;
    const int col = tn * 16 + ml;
    const size_t idx = (size_t)row * Nn + col;
    out[2 * idx + 0] = (c1[r] - c2[r]) + bre[col];
    out[2 * idx + 1] = (c3[r] + c4[r]) + bim[col];
  }
}

// ---------------------------------------------------------------------------
// Flash complex attention. One (b,h, 16-query) tile per wave.
// scores: s = q @ conj(k)^T * SCALE ; a = softmax(|s|) ; p = a * s/|s|
// O += p @ V  (complex), V fragments from the transposed wT copy.
// ---------------------------------------------------------------------------
__global__ __launch_bounds__(256) void attn_flash_kernel(
    const __bf16* __restrict__ wre, const __bf16* __restrict__ wim,
    const __bf16* __restrict__ wreT, const __bf16* __restrict__ wimT,
    __bf16* __restrict__ ore_out, __bf16* __restrict__ oim_out) {
  __shared__ __bf16 sPre[8][16 * 32];   // per-wave P tile (re), row-major
  __shared__ __bf16 sPim[8][16 * 32];   // per-wave P tile (im)

  const int lane = threadIdx.x & 31, wave = threadIdx.x >> 5;
  const int ml = lane & 15, g = lane >> 4;
  const int bh = blockIdx.y, b = bh / Hz, h = bh % Hz;
  const int q0 = blockIdx.x * 128 + wave * 16;
  const size_t headbase = (size_t)b * Nz * INNERz + (size_t)h * DHz;

  const __bf16* qre = wre + headbase + (size_t)(q0 + ml) * INNERz;
  const __bf16* qim = wim + headbase + (size_t)(q0 + ml) * INNERz;

  // Transposed-w pointers for V fragments: wT[(h*DH + d) * Mrows + b*N + key]
  // lane holds head-column d = ct*16 + ml; elements run contiguously over key.
  const __bf16* vTre[4];
  const __bf16* vTim[4];
#pragma unroll
  for (int ct = 0; ct < 4; ++ct) {
    const size_t vb = (size_t)(h * DHz + ct * 16 + ml) * Mrows + (size_t)b * Nz;
    vTre[ct] = wreT + vb;
    vTim[ct] = wimT + vb;
  }

  v8f ore[4], oim[4];
#pragma unroll
  for (int ct = 0; ct < 4; ++ct) { v8f z = {}; ore[ct] = z; oim[ct] = z; }
  float mrow[8], lrow[8];
#pragma unroll
  for (int r = 0; r < 8; ++r) { mrow[r] = -3.0e38f; lrow[r] = 0.f; }

  __bf16* pre = &sPre[wave][0];
  __bf16* pim = &sPim[wave][0];

  for (int jj = 0; jj < Nz; jj += 32) {
    // ---- score tiles: 16 queries x 32 keys, complex --------------------
    v8f sre[2], sia[2], sib[2];
    { v8f z = {}; sre[0] = z; sre[1] = z; sia[0] = z; sia[1] = z;
      sib[0] = z; sib[1] = z; }
#pragma unroll
    for (int ks = 0; ks < DHz; ks += 32) {
      const v16bf a_r = ldfragA(qre + ks, g);
      const v16bf a_i = ldfragA(qim + ks, g);
#pragma unroll
      for (int t = 0; t < 2; ++t) {
        const size_t krow = headbase + (size_t)(jj + t * 16 + ml) * INNERz + ks;
        const v16bf k_r = ldfragB(wre + krow, g);
        const v16bf k_i = ldfragB(wim + krow, g);
        sre[t] = wmma_bf(a_r, k_r, sre[t]);  // re: qr*kr + qi*ki
        sre[t] = wmma_bf(a_i, k_i, sre[t]);
        sia[t] = wmma_bf(a_i, k_r, sia[t]);  // im: qi*kr - qr*ki
        sib[t] = wmma_bf(a_r, k_i, sib[t]);
      }
    }

    // previous iteration's P-fragment LDS reads must be done before rewrite
    asm volatile("s_wait_dscnt 0" ::: "memory");

    // ---- magnitude softmax (flash update) ------------------------------
#pragma unroll
    for (int r = 0; r < 8; ++r) {
      const float er0 = sre[0][r] * SCALEz;
      const float ei0 = (sia[0][r] - sib[0][r]) * SCALEz;
      const float er1 = sre[1][r] * SCALEz;
      const float ei1 = (sia[1][r] - sib[1][r]) * SCALEz;
      const float mag0 = sqrtf(er0 * er0 + ei0 * ei0);
      const float mag1 = sqrtf(er1 * er1 + ei1 * ei1);
      const float mx = red_max16(fmaxf(mag0, mag1));
      const float mnew = fmaxf(mrow[r], mx);
      const float corr = __expf(mrow[r] - mnew);
      const float p0 = __expf(mag0 - mnew);
      const float p1 = __expf(mag1 - mnew);
      const float rs = red_sum16(p0 + p1);
      lrow[r] = lrow[r] * corr + rs;
      mrow[r] = mnew;
#pragma unroll
      for (int ct = 0; ct < 4; ++ct) { ore[ct][r] *= corr; oim[ct][r] *= corr; }

      float u0r = 1.f, u0i = 0.f, u1r = 1.f, u1i = 0.f;
      if (mag0 > 0.f) { u0r = er0 / mag0; u0i = ei0 / mag0; }
      if (mag1 > 0.f) { u1r = er1 / mag1; u1i = ei1 / mag1; }

      const int rowi = (r + 8 * g) * 32;   // tile row = r + 8*(lane/16)
      pre[rowi + ml]      = (__bf16)(p0 * u0r);
      pim[rowi + ml]      = (__bf16)(p0 * u0i);
      pre[rowi + 16 + ml] = (__bf16)(p1 * u1r);
      pim[rowi + 16 + ml] = (__bf16)(p1 * u1i);
    }

    // P stores visible before A-layout re-read (same wave, private tile)
    asm volatile("s_wait_dscnt 0" ::: "memory");

    const v16bf p_re  = ldfragA(pre + ml * 32, g);
    const v16bf p_im  = ldfragA(pim + ml * 32, g);
    const v16bf p_imn = negbf(p_im);

    // ---- O += P @ V (complex), V fragments from wT as two 16B loads per lane ------
#pragma unroll
    for (int ct = 0; ct < 4; ++ct) {
      const v16bf v_r = ldfragB(vTre[ct] + jj, g);
      const v16bf v_i = ldfragB(vTim[ct] + jj, g);
      ore[ct] = wmma_bf(p_re, v_r, ore[ct]);   // re: pr*vr - pi*vi
      ore[ct] = wmma_bf(p_imn, v_i, ore[ct]);
      oim[ct] = wmma_bf(p_re, v_i, oim[ct]);   // im: pr*vi + pi*vr
      oim[ct] = wmma_bf(p_im, v_r, oim[ct]);
    }
  }

  // ---- normalize and store (bf16, [B*N, INNER] layout) -----------------
#pragma unroll
  for (int r = 0; r < 8; ++r) {
    const float inv = 1.f / lrow[r];
    const int row = q0 + r + 8 * g;
#pragma unroll
    for (int ct = 0; ct < 4; ++ct) {
      const size_t idx = headbase + (size_t)row * INNERz + ct * 16 + ml;
      ore_out[idx] = (__bf16)(ore[ct][r] * inv);
      oim_out[idx] = (__bf16)(oim[ct][r] * inv);
    }
  }
}

// ---------------------------------------------------------------------------
// Host launcher
// ---------------------------------------------------------------------------
extern "C" void kernel_launch(void* const* d_in, const int* in_sizes, int n_in,
                              void* d_out, int out_size, void* d_ws,
                              size_t ws_size, hipStream_t stream) {
  const float* x_re    = (const float*)d_in[0];
  const float* x_im    = (const float*)d_in[1];
  const float* wqkv_re = (const float*)d_in[2];
  const float* wqkv_im = (const float*)d_in[3];
  const float* wout_re = (const float*)d_in[4];
  const float* wout_im = (const float*)d_in[5];
  const float* bout_re = (const float*)d_in[6];
  const float* bout_im = (const float*)d_in[7];

  constexpr size_t BND = (size_t)Bz * Nz * Dz;   // 4M elems
  constexpr size_t WSZ = (size_t)INNERz * Dz;    // 1M elems

  char* ws = (char*)d_ws;
  size_t off = 0;
  auto carve = [&](size_t elems) {
    void* p = ws + off;
    off += (elems * sizeof(__bf16) + 255) & ~(size_t)255;
    return (__bf16*)p;
  };
  __bf16* xr  = carve(BND);   // x_re bf16
  __bf16* xi  = carve(BND);   // x_im bf16
  __bf16* qr  = carve(WSZ);   // Wqkv re bf16
  __bf16* qi  = carve(WSZ);   // Wqkv im bf16
  __bf16* orw = carve(WSZ);   // Wout re bf16
  __bf16* oiw = carve(WSZ);   // Wout im bf16
  __bf16* wr  = carve(BND);   // w = x @ Wqkv^T, re (row-major [B*N, INNER])
  __bf16* wi  = carve(BND);   // w im
  __bf16* wrt = carve(BND);   // w re transposed [INNER, B*N]
  __bf16* wit = carve(BND);   // w im transposed
  __bf16* ar  = carve(BND);   // attention out re
  __bf16* ai  = carve(BND);   // attention out im

  // 1) fp32 -> bf16 conversions
  cvt_f32_bf16_kernel<<<1024, 256, 0, stream>>>(x_re, xr, (int)BND);
  cvt_f32_bf16_kernel<<<1024, 256, 0, stream>>>(x_im, xi, (int)BND);
  cvt_f32_bf16_kernel<<<512, 256, 0, stream>>>(wqkv_re, qr, (int)WSZ);
  cvt_f32_bf16_kernel<<<512, 256, 0, stream>>>(wqkv_im, qi, (int)WSZ);
  cvt_f32_bf16_kernel<<<512, 256, 0, stream>>>(wout_re, orw, (int)WSZ);
  cvt_f32_bf16_kernel<<<512, 256, 0, stream>>>(wout_im, oiw, (int)WSZ);

  // 2) QKV projection: w[4096,1024] = x @ Wqkv^T (complex), + transposed copy
  const int tiles = (Mrows / 16) * (INNERz / 16);   // 16384
  gemm_cplx_bf16_kernel<<<tiles / 8, 256, 0, stream>>>(
      xr, xi, qr, qi, wr, wi, wrt, wit, Mrows, INNERz, Dz);

  // 3) flash complex attention per (b,h)
  attn_flash_kernel<<<dim3(Nz / 128, Bz * Hz), dim3(256), 0, stream>>>(
      wr, wi, wrt, wit, ar, ai);

  // 4) output projection + bias -> interleaved complex64
  gemm_cplx_out_kernel<<<tiles / 8, 256, 0, stream>>>(
      ar, ai, orw, oiw, bout_re, bout_im, (float*)d_out, Mrows, Dz, INNERz);
}
